// PointTransformerBlock_45578192945247
// MI455X (gfx1250) — matrix operand order repaired and checked by reading, not compiled, not run on hardware
//
#include <hip/hip_runtime.h>
#include <hip/hip_bf16.h>

typedef _Float16 f16;
typedef __attribute__((ext_vector_type(16))) _Float16 v16h;
typedef __attribute__((ext_vector_type(8)))  float    v8f;

#define N_NODES_C 50000
#define N_EDGES_C 800000
#define E_TOT_C   (N_EDGES_C + N_NODES_C)   /* 850000, = 16 * 53125 */
#define N_ETILES  (E_TOT_C / 16)            /* 53125 */
#define N_NTILES  (N_NODES_C / 16)          /* 3125  */
#define DCH 64

// CDNA5 split-counter LDS fence: orders cross-lane LDS RAW within a wave.
__device__ __forceinline__ void lds_fence() {
  asm volatile("s_wait_dscnt 0" ::: "memory");
}

// One 16x16 output tile of a 16x64 @ 64x16 GEMM via two K=32 WMMAs.
// bp points at transposed-f16 weights wT[n][k] offset to (n_global*64 + hi*16).
__device__ __forceinline__ v8f gemm_tile(v16h a0, v16h a1, const f16* bp) {
  v16h b0 = *(const v16h*)(bp);        // K = 0..31 chunk (this lane's 16 halves)
  v16h b1 = *(const v16h*)(bp + 32);   // K = 32..63 chunk
  v8f c = {};
  c = __builtin_amdgcn_wmma_f32_16x16x32_f16(false, a0, false, b0, (short)0, c, false, false);
  c = __builtin_amdgcn_wmma_f32_16x16x32_f16(false, a1, false, b1, (short)0, c, false, false);
  return c;
}

// ---------------------------------------------------------------------------
// K1: x_val = x@W_lin, a_src = x@W_src, a_dst = x@W_dst  (16-row tiles / wave)
// ---------------------------------------------------------------------------
__global__ void __launch_bounds__(128) node_proj_kernel(
    const float* __restrict__ x,
    const float* __restrict__ Wl, const float* __restrict__ Ws, const float* __restrict__ Wd,
    float* __restrict__ xv, float* __restrict__ asr, float* __restrict__ ads)
{
  __shared__ alignas(32) f16 wL[64*64];
  __shared__ alignas(32) f16 wS[64*64];
  __shared__ alignas(32) f16 wD[64*64];
  int tid = threadIdx.x;
  for (int i = tid; i < 64*64; i += 128) {      // W[k][n] -> wT[n][k] in f16
    int k = i >> 6, n = i & 63;
    wL[n*64+k] = (f16)Wl[i];
    wS[n*64+k] = (f16)Ws[i];
    wD[n*64+k] = (f16)Wd[i];
  }
  __syncthreads();

  int wave = tid >> 5, lane = tid & 31;
  int m16 = lane & 15, hi = lane >> 4;
  int gw = blockIdx.x*4 + wave, nw = gridDim.x*4;
  for (int t = gw; t < N_NTILES; t += nw) {
    const float* xrow = x + (size_t)(t*16 + m16)*DCH;
    v16h a0, a1;
#pragma unroll
    for (int e = 0; e < 16; e++) {
      int k0 = (hi<<4) + e;
      a0[e] = (f16)xrow[k0];
      a1[e] = (f16)xrow[k0+32];
    }
#pragma unroll
    for (int nt = 0; nt < 4; nt++) {
      int n = nt*16 + m16;
      v8f c;
      c = gemm_tile(a0, a1, wL + n*64 + (hi<<4));
#pragma unroll
      for (int r = 0; r < 8; r++) xv [(size_t)(t*16 + r + (hi<<3))*DCH + n] = c[r];
      c = gemm_tile(a0, a1, wS + n*64 + (hi<<4));
#pragma unroll
      for (int r = 0; r < 8; r++) asr[(size_t)(t*16 + r + (hi<<3))*DCH + n] = c[r];
      c = gemm_tile(a0, a1, wD + n*64 + (hi<<4));
#pragma unroll
      for (int r = 0; r < 8; r++) ads[(size_t)(t*16 + r + (hi<<3))*DCH + n] = c[r];
    }
  }
}

// ---------------------------------------------------------------------------
// K2: zero z, out, stats
// ---------------------------------------------------------------------------
__global__ void zero_kernel(float* __restrict__ z, float* __restrict__ o,
                            float* __restrict__ stats, int total)
{
  int i = blockIdx.x*blockDim.x + threadIdx.x;
  if (i < total) { z[i] = 0.f; o[i] = 0.f; }
  if (i < 128) stats[i] = 0.f;
}

// ---------------------------------------------------------------------------
// Edge passes. PASS==1: z[dst,ch] += exp(alpha). PASS==2: out[dst,ch] +=
// exp(alpha)/(z+1e-16) * (x_val[src,ch] + delta). Softmax max-shift omitted
// (shift-invariant; logits are tiny with 0.05-scale weights).
// ---------------------------------------------------------------------------
template<int PASS>
__global__ void __launch_bounds__(128) edge_kernel(
    const float* __restrict__ pos, const int* __restrict__ ei,
    const float* __restrict__ xv, const float* __restrict__ asr, const float* __restrict__ ads,
    const float* __restrict__ pw1, const float* __restrict__ pb1,
    const float* __restrict__ pw2, const float* __restrict__ pb2,
    const float* __restrict__ aw1, const float* __restrict__ ab1,
    const float* __restrict__ aw2, const float* __restrict__ ab2,
    float* __restrict__ zbuf, float* __restrict__ outbuf)
{
  __shared__ alignas(32) f16 wP2[64*64];   // transposed f16 weights [n][k]
  __shared__ alignas(32) f16 wA1[64*64];
  __shared__ alignas(32) f16 wA2[64*64];
  __shared__ float sPW1[3*64];
  __shared__ float sPB1[64], sPB2[64], sAB1[64], sAB2[64];
  __shared__ float sDelta[4][16*65];       // per-wave, stride 65: no bank conflicts
  __shared__ float sHid  [4][16*65];
  __shared__ float sRel  [4][48];
  __shared__ int   sSrc  [4][16], sDst[4][16];

  int tid = threadIdx.x;
  for (int i = tid; i < 64*64; i += 128) {
    int k = i >> 6, n = i & 63;
    wP2[n*64+k] = (f16)pw2[i];
    wA1[n*64+k] = (f16)aw1[i];
    wA2[n*64+k] = (f16)aw2[i];
  }
  for (int i = tid; i < 192; i += 128) sPW1[i] = pw1[i];
  if (tid < 64) { sPB1[tid]=pb1[tid]; sPB2[tid]=pb2[tid]; sAB1[tid]=ab1[tid]; sAB2[tid]=ab2[tid]; }
  __syncthreads();

  int wave = tid >> 5, lane = tid & 31;
  int m16 = lane & 15, hi = lane >> 4;
  float* dbuf = sDelta[wave];
  float* hbuf = sHid[wave];
  int gw = blockIdx.x*4 + wave, nw = gridDim.x*4;

  for (int t = gw; t < N_ETILES; t += nw) {
    // --- gather tile indices and relative positions (16 edges) -------------
    if (lane < 16) {
      int ee = t*16 + lane;
      int s, d;
      if (ee < N_EDGES_C) { s = ei[ee]; d = ei[N_EDGES_C + ee]; }
      else                { s = ee - N_EDGES_C; d = s; }          // self loop
      sSrc[wave][lane] = s; sDst[wave][lane] = d;
      sRel[wave][lane*3+0] = pos[(size_t)d*3+0] - pos[(size_t)s*3+0];
      sRel[wave][lane*3+1] = pos[(size_t)d*3+1] - pos[(size_t)s*3+1];
      sRel[wave][lane*3+2] = pos[(size_t)d*3+2] - pos[(size_t)s*3+2];
    }
    lds_fence();

    // --- pos MLP layer 1 (K=3, pure VALU) directly into A-fragments --------
    float r0 = sRel[wave][m16*3+0];
    float r1 = sRel[wave][m16*3+1];
    float r2 = sRel[wave][m16*3+2];
    v16h aP0, aP1;
#pragma unroll
    for (int e = 0; e < 16; e++) {
      int k0 = (hi<<4) + e;
      int k1 = k0 + 32;
      aP0[e] = (f16)(r0*sPW1[k0] + r1*sPW1[64+k0] + r2*sPW1[128+k0] + sPB1[k0]);
      aP1[e] = (f16)(r0*sPW1[k1] + r1*sPW1[64+k1] + r2*sPW1[128+k1] + sPB1[k1]);
    }
    // --- pos MLP layer 2 (WMMA): delta -> LDS ------------------------------
#pragma unroll
    for (int nt = 0; nt < 4; nt++) {
      int n = nt*16 + m16;
      v8f c = gemm_tile(aP0, aP1, wP2 + n*64 + (hi<<4));
      float bias = sPB2[n];
#pragma unroll
      for (int r = 0; r < 8; r++) dbuf[(r + (hi<<3))*65 + n] = c[r] + bias;
    }
    lds_fence();

    // --- alpha_in = a_dst[dst] - a_src[src] + delta  -> A fragments --------
    {
      int sj = sSrc[wave][m16], dj = sDst[wave][m16];
      const float* adRow = ads + (size_t)dj*DCH;
      const float* asRow = asr + (size_t)sj*DCH;
      v16h aA0, aA1;
#pragma unroll
      for (int e = 0; e < 16; e++) {
        int k0 = (hi<<4) + e;
        int k1 = k0 + 32;
        aA0[e] = (f16)(adRow[k0] - asRow[k0] + dbuf[m16*65 + k0]);
        aA1[e] = (f16)(adRow[k1] - asRow[k1] + dbuf[m16*65 + k1]);
      }
      // --- attn MLP layer 1 (WMMA): hid -> LDS -----------------------------
#pragma unroll
      for (int nt = 0; nt < 4; nt++) {
        int n = nt*16 + m16;
        v8f c = gemm_tile(aA0, aA1, wA1 + n*64 + (hi<<4));
        float bias = sAB1[n];
#pragma unroll
        for (int r = 0; r < 8; r++) hbuf[(r + (hi<<3))*65 + n] = c[r] + bias;
      }
    }
    lds_fence();

    // --- attn MLP layer 2 (WMMA) + softmax/aggregate epilogue --------------
    v16h aH0, aH1;
#pragma unroll
    for (int e = 0; e < 16; e++) {
      int k0 = (hi<<4) + e;
      aH0[e] = (f16)hbuf[m16*65 + k0];
      aH1[e] = (f16)hbuf[m16*65 + k0 + 32];
    }
#pragma unroll
    for (int nt = 0; nt < 4; nt++) {
      int n = nt*16 + m16;
      v8f c = gemm_tile(aH0, aH1, wA2 + n*64 + (hi<<4));
      float bias = sAB2[n];
#pragma unroll
      for (int r = 0; r < 8; r++) {
        int   mm = r + (hi<<3);
        int   dd = sDst[wave][mm];
        float ev = __expf(c[r] + bias);
        size_t o = (size_t)dd*DCH + n;
        if (PASS == 1) {
          atomicAdd(&zbuf[o], ev);                           // denominator
        } else {
          float w   = ev / (zbuf[o] + 1e-16f);               // softmax weight
          int   ss  = sSrc[wave][mm];
          float msg = w * (xv[(size_t)ss*DCH + n] + dbuf[mm*65 + n]);
          atomicAdd(&outbuf[o], msg);                        // scatter-sum
        }
      }
    }
  }
}

// ---------------------------------------------------------------------------
// ELU in place + per-channel sum / sumsq (LDS-reduced, then global atomics)
// ---------------------------------------------------------------------------
__global__ void __launch_bounds__(256) elu_stats_kernel(
    float* __restrict__ out, float* __restrict__ stats, int total)
{
  __shared__ float ssum[64], ssq[64];
  int tid = threadIdx.x;
  if (tid < 64) { ssum[tid] = 0.f; ssq[tid] = 0.f; }
  __syncthreads();
  int i = blockIdx.x*256 + tid;
  if (i < total) {
    float v = out[i];
    v = v > 0.f ? v : (__expf(v) - 1.f);    // ELU (alpha=1)
    out[i] = v;
    int ch = i & 63;
    atomicAdd(&ssum[ch], v);
    atomicAdd(&ssq[ch],  v*v);
  }
  __syncthreads();
  if (tid < 64) {
    atomicAdd(&stats[tid],      ssum[tid]);
    atomicAdd(&stats[64 + tid], ssq[tid]);
  }
}

__global__ void bn_finalize_kernel(const float* __restrict__ stats,
                                   const float* __restrict__ gamma,
                                   const float* __restrict__ beta,
                                   float* __restrict__ ss)
{
  int c = threadIdx.x;
  if (c < 64) {
    float mean = stats[c] * (1.0f / (float)N_NODES_C);
    float var  = stats[64 + c] * (1.0f / (float)N_NODES_C) - mean*mean;
    float rstd = rsqrtf(var + 1e-5f);
    float sc   = rstd * gamma[c];
    ss[c]      = sc;
    ss[64 + c] = beta[c] - mean*sc;
  }
}

__global__ void bn_apply_kernel(float* __restrict__ out,
                                const float* __restrict__ ss, int total)
{
  int i = blockIdx.x*blockDim.x + threadIdx.x;
  if (i < total) {
    int ch = i & 63;
    out[i] = out[i]*ss[ch] + ss[64 + ch];
  }
}

// ---------------------------------------------------------------------------
extern "C" void kernel_launch(void* const* d_in, const int* in_sizes, int n_in,
                              void* d_out, int out_size, void* d_ws, size_t ws_size,
                              hipStream_t stream) {
  (void)in_sizes; (void)n_in; (void)out_size; (void)ws_size;
  const float* x    = (const float*)d_in[0];
  const float* pos  = (const float*)d_in[1];
  const int*   ei   = (const int*)  d_in[2];
  const float* Wl   = (const float*)d_in[3];
  const float* Wsrc = (const float*)d_in[4];
  const float* Wdst = (const float*)d_in[5];
  const float* pw1  = (const float*)d_in[6];
  const float* pb1  = (const float*)d_in[7];
  const float* pw2  = (const float*)d_in[8];
  const float* pb2  = (const float*)d_in[9];
  const float* aw1  = (const float*)d_in[10];
  const float* ab1  = (const float*)d_in[11];
  const float* aw2  = (const float*)d_in[12];
  const float* ab2  = (const float*)d_in[13];
  const float* gamma= (const float*)d_in[14];
  const float* beta = (const float*)d_in[15];
  float* out = (float*)d_out;

  const size_t NDf = (size_t)N_NODES_C * DCH;   // 3.2M floats
  float* ws    = (float*)d_ws;                  // ~51.2 MB used total
  float* xv    = ws;
  float* asr   = ws + NDf;
  float* ads   = ws + 2*NDf;
  float* zb    = ws + 3*NDf;
  float* stats = ws + 4*NDf;                    // sum[64], sumsq[64]
  float* ss    = stats + 128;                   // scale[64], shift[64]

  const int total = (int)NDf;
  node_proj_kernel<<<512, 128, 0, stream>>>(x, Wl, Wsrc, Wdst, xv, asr, ads);
  zero_kernel<<<(total + 255)/256, 256, 0, stream>>>(zb, out, stats, total);
  edge_kernel<1><<<2048, 128, 0, stream>>>(pos, ei, xv, asr, ads,
                                           pw1, pb1, pw2, pb2,
                                           aw1, ab1, aw2, ab2, zb, out);
  edge_kernel<2><<<2048, 128, 0, stream>>>(pos, ei, xv, asr, ads,
                                           pw1, pb1, pw2, pb2,
                                           aw1, ab1, aw2, ab2, zb, out);
  elu_stats_kernel<<<(total + 255)/256, 256, 0, stream>>>(out, stats, total);
  bn_finalize_kernel<<<1, 64, 0, stream>>>(stats, gamma, beta, ss);
  bn_apply_kernel<<<(total + 255)/256, 256, 0, stream>>>(out, ss, total);
}
